// MultiHeadedSelfAttention_54966991454290
// MI455X (gfx1250) — matrix operand order repaired and checked by reading, not compiled, not run on hardware
//
#include <hip/hip_runtime.h>
#include <hip/hip_bf16.h>

// MHA forward, MI455X/gfx1250, wave32, v_wmma_f32_16x16x32_f16 everywhere.
// B=2, S=2048, D=1024, H=16, W=64.
// Register-blocked: 32 queries/wave (attention), 32 rows/wave (QKV GEMM) to
// amortize operand loads over 2x the WMMAs (loads are the issue bottleneck;
// data is L2-resident so HBM is not the limiter).

#define BB 2
#define SS 2048
#define DD 1024
#define HH 16
#define WW 64

typedef _Float16 half8  __attribute__((ext_vector_type(8)));
typedef _Float16 half16 __attribute__((ext_vector_type(16)));
typedef float    f32x8  __attribute__((ext_vector_type(8)));

static __device__ __forceinline__ f32x8 zero8() {
    f32x8 z;
#pragma unroll
    for (int i = 0; i < 8; ++i) z[i] = 0.0f;
    return z;
}

// A-fragment (16x32 f16): lane half hf holds K = {hf*8..hf*8+7, 16+hf*8..16+hf*8+7}
// p points at (row, k0 + hf*8); second group is +16 elements (two b128 loads).
static __device__ __forceinline__ half16 load_a_frag(const _Float16* __restrict__ p) {
    half8 lo = *(const half8*)p;
    half8 hi = *(const half8*)(p + 16);
    half16 r;
#pragma unroll
    for (int i = 0; i < 8; ++i) { r[i] = lo[i]; r[i + 8] = hi[i]; }
    return r;
}

// B-fragment (32x16 f16): lane = column N, half hf holds K = {hf*8..hf*8+7, 16+hf*8..16+hf*8+7}.
static __device__ __forceinline__ half16 load_b_frag(const _Float16* __restrict__ p) {
    return load_a_frag(p);
}

static __device__ __forceinline__ f32x8 wmma_f16(half16 a, half16 b, f32x8 c) {
    return __builtin_amdgcn_wmma_f32_16x16x32_f16(false, a, false, b, (short)0, c, false, false);
}

// ---------------- prep: f32 -> f16 conversions ----------------

__global__ void cvt_x_kernel(const float* __restrict__ x, _Float16* __restrict__ xh, int n) {
    int i = blockIdx.x * blockDim.x + threadIdx.x;
    if (i < n) xh[i] = (_Float16)x[i];
}

// wt[z][o][i] = W_z[i][o]  (transpose so B-fragments read contiguous K)
__global__ void transpose_w_kernel(const float* __restrict__ Wq, const float* __restrict__ Wk,
                                   const float* __restrict__ Wv, _Float16* __restrict__ wt, int n) {
    int idx = blockIdx.x * blockDim.x + threadIdx.x;
    if (idx >= n) return;
    int z = idx >> 20;            // / (1024*1024)
    int rem = idx & 1048575;
    int o = rem >> 10;
    int i = rem & 1023;
    const float* W = (z == 0) ? Wq : (z == 1) ? Wk : Wv;
    wt[idx] = (_Float16)W[i * DD + o];
}

// ---------------- QKV projection GEMM ----------------
// grid: (32, 16, 3), block 128 (4 waves). wave -> 32 rows x 64 cols output tile
// (two 16-row A tiles share every W B-fragment -> half the B loads per WMMA).
// z=0: Q (scaled by 1/8), z=1: K, z=2: V stored transposed [b,h,w,s].

__global__ __launch_bounds__(128) void qkv_gemm_kernel(
    const _Float16* __restrict__ xh, const _Float16* __restrict__ wt,
    const float* __restrict__ bq, const float* __restrict__ bk, const float* __restrict__ bv,
    _Float16* __restrict__ Qh, _Float16* __restrict__ Kh, _Float16* __restrict__ Vt) {
    const int lane = threadIdx.x & 31;
    const int wave = threadIdx.x >> 5;
    const int l = lane & 15, hf = lane >> 4;
    const int z = blockIdx.z;
    const int m0 = (blockIdx.x * 4 + wave) * 32;
    const int h = blockIdx.y;
    const int n0 = h * 64;
    const _Float16* w = wt + (size_t)z * DD * DD;
    const float* bias = (z == 0) ? bq : (z == 1) ? bk : bv;

    f32x8 acc[2][4];
#pragma unroll
    for (int t = 0; t < 2; ++t)
#pragma unroll
        for (int j = 0; j < 4; ++j) acc[t][j] = zero8();

    const _Float16* arow0 = xh + (size_t)(m0 + l) * DD;
    const _Float16* arow1 = xh + (size_t)(m0 + 16 + l) * DD;
    for (int k0 = 0; k0 < DD; k0 += 32) {
        half16 a0 = load_a_frag(arow0 + k0 + hf * 8);
        half16 a1 = load_a_frag(arow1 + k0 + hf * 8);
#pragma unroll
        for (int j = 0; j < 4; ++j) {
            half16 bf = load_b_frag(w + (size_t)(n0 + j * 16 + l) * DD + k0 + hf * 8);
            acc[0][j] = wmma_f16(a0, bf, acc[0][j]);
            acc[1][j] = wmma_f16(a1, bf, acc[1][j]);
        }
    }

    const int bidx = m0 >> 11;     // / S
    const int s0 = m0 & (SS - 1);
#pragma unroll
    for (int t = 0; t < 2; ++t) {
#pragma unroll
        for (int j = 0; j < 4; ++j) {
            const int n = n0 + j * 16 + l;
            const float bn = bias[n];
#pragma unroll
            for (int r = 0; r < 8; ++r) {
                float v = acc[t][j][r] + bn;
                if (z == 0) v *= 0.125f;          // 1/sqrt(W) folded into Q
                const int srow = s0 + t * 16 + r + hf * 8;
                const _Float16 hv = (_Float16)v;
                if (z == 2) {
                    Vt[(((size_t)bidx * HH + h) * WW + j * 16 + l) * SS + srow] = hv;
                } else {
                    _Float16* dst = (z == 0) ? Qh : Kh;
                    dst[(((size_t)bidx * HH + h) * SS + srow) * WW + j * 16 + l] = hv;
                }
            }
        }
    }
}

// ---------------- flash attention ----------------
// grid: (16, 16, 2), block 128 (4 waves). wave -> 32-query tile (two 16-query
// subtiles), streams 32 keys/iter. K/V fragments feed both subtiles -> 16 WMMA
// per 16 b128 loads per iter. Computes S^T = K * Q^T so softmax stats are
// lane-local and P maps lane-locally into the A-fragment for P*V.

__global__ __launch_bounds__(128) void attn_kernel(
    const _Float16* __restrict__ Qh, const _Float16* __restrict__ Kh,
    const _Float16* __restrict__ Vt, float* __restrict__ out) {
    const int lane = threadIdx.x & 31;
    const int wave = threadIdx.x >> 5;
    const int l = lane & 15, hf = lane >> 4;
    const int sq = (blockIdx.x * 4 + wave) * 32;
    const int h = blockIdx.y;
    const int b = blockIdx.z;
    const int bh = b * HH + h;

    const _Float16* Qb = Qh + (size_t)bh * SS * WW;
    const _Float16* Kb = Kh + (size_t)bh * SS * WW;
    const _Float16* Vb = Vt + (size_t)bh * WW * SS;

    // Q as B-fragments of Q^T (columns = queries): lane l -> query sq+t*16+l,
    // K-dim = features, chunk c covers features c*32 .. c*32+31.
    half16 qf[2][2];
#pragma unroll
    for (int t = 0; t < 2; ++t)
#pragma unroll
        for (int c = 0; c < 2; ++c)
            qf[t][c] = load_b_frag(Qb + (size_t)(sq + t * 16 + l) * WW + c * 32 + hf * 8);

    f32x8 o[2][4];
#pragma unroll
    for (int t = 0; t < 2; ++t)
#pragma unroll
        for (int j = 0; j < 4; ++j) o[t][j] = zero8();
    float mrow[2] = {-__builtin_inff(), -__builtin_inff()};
    float lrow[2] = {0.0f, 0.0f};

    for (int kb = 0; kb < SS; kb += 32) {
        // S^T tiles per q-subtile t: st[t][0] = keys kb..kb+15 (rows) x queries,
        // st[t][1] = keys kb+16..kb+31.
        f32x8 st[2][2];
#pragma unroll
        for (int t = 0; t < 2; ++t) { st[t][0] = zero8(); st[t][1] = zero8(); }
#pragma unroll
        for (int c = 0; c < 2; ++c) {
            half16 ka0 = load_a_frag(Kb + (size_t)(kb + l) * WW + c * 32 + hf * 8);
            half16 ka1 = load_a_frag(Kb + (size_t)(kb + 16 + l) * WW + c * 32 + hf * 8);
#pragma unroll
            for (int t = 0; t < 2; ++t) {
                st[t][0] = wmma_f16(ka0, qf[t][c], st[t][0]);
                st[t][1] = wmma_f16(ka1, qf[t][c], st[t][1]);
            }
        }

        // Online softmax per subtile: each lane holds 16 scores of query
        // (sq + t*16 + l); halves hold complementary keys -> one shfl_xor(16).
        half16 pa[2];
        float corr[2];
#pragma unroll
        for (int t = 0; t < 2; ++t) {
            float vmax = st[t][0][0];
#pragma unroll
            for (int r = 1; r < 8; ++r) vmax = fmaxf(vmax, st[t][0][r]);
#pragma unroll
            for (int r = 0; r < 8; ++r) vmax = fmaxf(vmax, st[t][1][r]);
            vmax = fmaxf(vmax, __shfl_xor(vmax, 16, 32));
            const float mnew = fmaxf(mrow[t], vmax);
            corr[t] = __expf(mrow[t] - mnew);
            float psum = 0.0f;
#pragma unroll
            for (int r = 0; r < 8; ++r) {
                const float p0 = __expf(st[t][0][r] - mnew);
                const float p1 = __expf(st[t][1][r] - mnew);
                psum += p0 + p1;
                pa[t][r] = (_Float16)p0;       // keys hf*8 .. +7
                pa[t][r + 8] = (_Float16)p1;   // keys 16+hf*8 .. +7
            }
            psum += __shfl_xor(psum, 16, 32);
            lrow[t] = lrow[t] * corr[t] + psum;
            mrow[t] = mnew;
        }

        // Rescale output accumulators: C-row r lives at VGPR r (rows 0-7 in
        // lanes 0-15, rows 8-15 in lanes 16-31); broadcast factor via bpermute.
#pragma unroll
        for (int t = 0; t < 2; ++t)
#pragma unroll
            for (int r = 0; r < 8; ++r) {
                const int src = (lane & 16) ? (8 + r) : r;
                const float f = __shfl(corr[t], src, 32);
#pragma unroll
                for (int j = 0; j < 4; ++j) o[t][j][r] *= f;
            }

        // O += P * V ; each V B-fragment feeds both q-subtiles.
#pragma unroll
        for (int j = 0; j < 4; ++j) {
            half16 vb = load_b_frag(Vb + (size_t)(j * 16 + l) * SS + kb + hf * 8);
#pragma unroll
            for (int t = 0; t < 2; ++t) o[t][j] = wmma_f16(pa[t], vb, o[t][j]);
        }
    }

#pragma unroll
    for (int t = 0; t < 2; ++t) {
        const float inv = 1.0f / lrow[t];
#pragma unroll
        for (int r = 0; r < 8; ++r) {
            const int src = (lane & 16) ? (8 + r) : r;
            const float f = __shfl(inv, src, 32);
            const int srow = sq + t * 16 + r + hf * 8;
#pragma unroll
            for (int j = 0; j < 4; ++j) {
                out[((size_t)b * SS + srow) * DD + h * WW + j * 16 + l] = o[t][j][r] * f;
            }
        }
    }
}

// ---------------- launcher ----------------

extern "C" void kernel_launch(void* const* d_in, const int* in_sizes, int n_in,
                              void* d_out, int out_size, void* d_ws, size_t ws_size,
                              hipStream_t stream) {
    const float* x  = (const float*)d_in[0];
    const float* Wq = (const float*)d_in[1];
    const float* bq = (const float*)d_in[2];
    const float* Wk = (const float*)d_in[3];
    const float* bk = (const float*)d_in[4];
    const float* Wv = (const float*)d_in[5];
    const float* bv = (const float*)d_in[6];
    float* out = (float*)d_out;

    char* ws = (char*)d_ws;
    const size_t XH_BYTES = (size_t)BB * SS * DD * sizeof(_Float16);      // 8 MB
    const size_t WT_BYTES = (size_t)3 * DD * DD * sizeof(_Float16);       // 6 MB
    const size_t QH_BYTES = (size_t)BB * HH * SS * WW * sizeof(_Float16); // 8 MB
    _Float16* xh = (_Float16*)(ws);
    _Float16* wt = (_Float16*)(ws + XH_BYTES);
    _Float16* Qh = (_Float16*)(ws + XH_BYTES + WT_BYTES);
    _Float16* Kh = (_Float16*)(ws + XH_BYTES + WT_BYTES + QH_BYTES);
    _Float16* Vt = (_Float16*)(ws + XH_BYTES + WT_BYTES + 2 * QH_BYTES);

    {
        const int n = BB * SS * DD;
        cvt_x_kernel<<<(n + 255) / 256, 256, 0, stream>>>(x, xh, n);
    }
    {
        const int n = 3 * DD * DD;
        transpose_w_kernel<<<(n + 255) / 256, 256, 0, stream>>>(Wq, Wk, Wv, wt, n);
    }
    // (B*S)/(32 rows * 4 waves) = 32 ; 16 head-tiles of 64 cols ; 3 projections
    qkv_gemm_kernel<<<dim3(32, 16, 3), 128, 0, stream>>>(xh, wt, bq, bk, bv, Qh, Kh, Vt);
    // S/(32 queries * 4 waves) = 16 ; H ; B
    attn_kernel<<<dim3(16, 16, 2), 128, 0, stream>>>(Qh, Kh, Vt, out);
}
